// MIDFUSION_IMG_TRANS_VSLT_GRUD_65420941853090
// MI455X (gfx1250) — matrix operand order repaired and checked by reading, not compiled, not run on hardware
//
#include <hip/hip_runtime.h>
#include <hip/hip_bf16.h>
#include <cstdint>

// ---------------- constants (all shapes are compile-time) ----------------
static constexpr int Bn = 32, Tn = 24, Fn = 18, Dn = 256;
static constexpr int SV = 196, SF = 221, NHh = 8, DHd = 32, FFn = 1024, NTk = 12;
static constexpr int SP = 224;                  // padded attention tile (mult of 32)
static constexpr int ROWPAD = 64;               // extra rows on q/k/v for edge-tile reads
static constexpr long long PER_BLK = 786432;    // bf16 elems per transformer block

typedef __attribute__((ext_vector_type(16))) __bf16 v16bf;
typedef __attribute__((ext_vector_type(8)))  float  v8f;

// ---------------- weight prep: f32 [K,N] -> bf16 transposed [N,K] --------
__global__ void wprep_kernel(const float* __restrict__ W, __bf16* __restrict__ Wt,
                             int K, int N) {
  int i = blockIdx.x * blockDim.x + threadIdx.x;
  if (i >= K * N) return;
  int k = i / N, n = i % N;
  Wt[(long long)n * K + k] = (__bf16)W[i];
}

__global__ void zero_kernel(float* __restrict__ p, int n) {
  int i = blockIdx.x * blockDim.x + threadIdx.x;
  if (i < n) p[i] = 0.0f;
}

// ---------------- generic WMMA GEMM ----------------
// C[M,N] = act( scale * (A[M,K] x B) + bias ) + resid
// BMODE 0: B = bf16 Wt[N,K] (pre-transposed weights)
// BMODE 1: B = f32 [N,K]  (NT: Q·K^T, B = K-matrix rows)
// BMODE 2: B = f32 [K,N]  (NN: attn·V)
// All loads are UNCONDITIONAL (buffers padded so edge-tile reads are safe);
// only stores are guarded. Each wave: one A frag, two N-tiles, two wmma.
template <int BMODE>
__global__ void __launch_bounds__(128) gemm_wmma_kernel(
    const float* __restrict__ A, const void* __restrict__ Bp,
    const float* __restrict__ bias, const float* __restrict__ resid,
    float* __restrict__ C, int M, int N, int K,
    int lda, int ldb, int ldc, int Hs,
    long long sAb, long long sAh, long long sBb, long long sBh,
    long long sCb, long long sCh, float scale, int act) {
  int wave = threadIdx.x >> 5;
  int lane = threadIdx.x & 31;
  int r    = lane & 15;
  int half = lane >> 4;
  int row0 = blockIdx.x * 16;
  int col0 = (blockIdx.y * 4 + wave) * 32;     // two 16-wide tiles per wave
  if (col0 >= N) return;                       // uniform per wave

  int bz = blockIdx.z;
  int bb = bz / Hs, hh = bz % Hs;
  const float* Ab = A + bb * sAb + hh * sAh;
  long long offB  = bb * sBb + hh * sBh;
  float* Cb       = C + bb * sCb + hh * sCh;
  const float* Rb = resid ? (resid + bb * sCb + hh * sCh) : nullptr;

  int mrow = row0 + r;
  int nc0  = col0 + r;
  int nc1  = col0 + 16 + r;

  v8f acc0 = {}, acc1 = {};
  for (int k0 = 0; k0 < K; k0 += 32) {
    // ---- A fragment: one contiguous 16-float run per half, vectorized ----
    const float* ap = Ab + (long long)mrow * lda + k0 + half * 16;
    float4 a0 = *(const float4*)(ap);
    float4 a1 = *(const float4*)(ap + 4);
    float4 a2 = *(const float4*)(ap + 8);
    float4 a3 = *(const float4*)(ap + 12);
    v16bf af;
    af[0]  = (__bf16)a0.x; af[1]  = (__bf16)a0.y; af[2]  = (__bf16)a0.z; af[3]  = (__bf16)a0.w;
    af[4]  = (__bf16)a1.x; af[5]  = (__bf16)a1.y; af[6]  = (__bf16)a1.z; af[7]  = (__bf16)a1.w;
    af[8]  = (__bf16)a2.x; af[9]  = (__bf16)a2.y; af[10] = (__bf16)a2.z; af[11] = (__bf16)a2.w;
    af[12] = (__bf16)a3.x; af[13] = (__bf16)a3.y; af[14] = (__bf16)a3.z; af[15] = (__bf16)a3.w;

    // ---- B fragments for the two N-tiles ----
    auto loadB = [&](int ncol) -> v16bf {
      if constexpr (BMODE == 0) {
        const __bf16* wp = (const __bf16*)Bp + offB + (long long)ncol * ldb + k0 + half * 16;
        return *(const v16bf*)wp;              // 32B contiguous -> 2x b128
      } else if constexpr (BMODE == 1) {
        const float* wp = (const float*)Bp + offB + (long long)ncol * ldb + k0 + half * 16;
        float4 f0 = *(const float4*)(wp);
        float4 f1 = *(const float4*)(wp + 4);
        float4 f2 = *(const float4*)(wp + 8);
        float4 f3 = *(const float4*)(wp + 12);
        v16bf o;
        o[0]  = (__bf16)f0.x; o[1]  = (__bf16)f0.y; o[2]  = (__bf16)f0.z; o[3]  = (__bf16)f0.w;
        o[4]  = (__bf16)f1.x; o[5]  = (__bf16)f1.y; o[6]  = (__bf16)f1.z; o[7]  = (__bf16)f1.w;
        o[8]  = (__bf16)f2.x; o[9]  = (__bf16)f2.y; o[10] = (__bf16)f2.z; o[11] = (__bf16)f2.w;
        o[12] = (__bf16)f3.x; o[13] = (__bf16)f3.y; o[14] = (__bf16)f3.z; o[15] = (__bf16)f3.w;
        return o;
      } else {
        const float* wp = (const float*)Bp + offB + (long long)(k0 + half * 16) * ldb + ncol;
        v16bf o;
#pragma unroll
        for (int j = 0; j < 16; ++j) o[j] = (__bf16)wp[(long long)j * ldb];
        return o;
      }
    };
    v16bf bf0 = loadB(nc0);
    v16bf bf1 = loadB(nc1);
    acc0 = __builtin_amdgcn_wmma_f32_16x16x32_bf16(false, af, false, bf0,
                                                   (short)0, acc0, false, false);
    acc1 = __builtin_amdgcn_wmma_f32_16x16x32_bf16(false, af, false, bf1,
                                                   (short)0, acc1, false, false);
  }

  auto storeT = [&](v8f& acc, int nc) {
#pragma unroll
    for (int i = 0; i < 8; ++i) {              // C: M = i + 8*half, N = lane%16
      int mr = row0 + half * 8 + i;
      if (mr < M && nc < N) {
        float v = acc[i] * scale;
        if (bias) v += bias[nc];
        if (act == 1) v = fmaxf(v, 0.0f);
        else if (act == 2) {
          float t = v;
          v = 0.5f * t * (1.0f + tanhf(0.79788456f * (t + 0.044715f * t * t * t)));
        }
        if (Rb) v += Rb[(long long)mr * ldc + nc];
        Cb[(long long)mr * ldc + nc] = v;
      }
    }
  };
  storeT(acc0, nc0);
  storeT(acc1, nc1);
}

// ---------------- layernorm over D=256, one block per row ----------------
__global__ void __launch_bounds__(256) layernorm256_kernel(
    const float* __restrict__ in, float* __restrict__ out,
    const float* __restrict__ g, const float* __restrict__ b) {
  __shared__ float red[256];
  long long row = blockIdx.x;
  int t = threadIdx.x;
  float v = in[row * 256 + t];
  red[t] = v; __syncthreads();
  for (int s = 128; s > 0; s >>= 1) { if (t < s) red[t] += red[t + s]; __syncthreads(); }
  float mu = red[0] * (1.0f / 256.0f); __syncthreads();
  float d = v - mu;
  red[t] = d * d; __syncthreads();
  for (int s = 128; s > 0; s >>= 1) { if (t < s) red[t] += red[t + s]; __syncthreads(); }
  float var = red[0] * (1.0f / 256.0f);
  out[row * 256 + t] = d * rsqrtf(var + 1e-5f) * g[t] + b[t];
}

// ---------------- masked softmax over keys (row stride SP=224) -----------
__global__ void __launch_bounds__(256) softmax_kernel(
    float* __restrict__ s, const float* __restrict__ mask, int H, int S) {
  __shared__ float red[256];
  int t  = threadIdx.x;
  int rq = blockIdx.x % S;
  int bz = blockIdx.x / S;
  int b  = bz / H;
  float* p = s + ((long long)bz * SP + rq) * SP;
  float v = -1e30f;
  if (t < S) { v = p[t]; if (mask) v += mask[b * S + t]; }
  red[t] = v; __syncthreads();
  for (int st = 128; st > 0; st >>= 1) { if (t < st) red[t] = fmaxf(red[t], red[t + st]); __syncthreads(); }
  float mx = red[0]; __syncthreads();
  float e = (t < S) ? __expf(v - mx) : 0.0f;
  red[t] = e; __syncthreads();
  for (int st = 128; st > 0; st >>= 1) { if (t < st) red[t] += red[t + st]; __syncthreads(); }
  float sum = red[0];
  if (t < S) p[t] = e / sum;
}

// ---------------- GRU-D input decay adjust ----------------
__global__ void xadj_kernel(const float* __restrict__ x, const float* __restrict__ m,
                            const float* __restrict__ dd, const float* __restrict__ w,
                            const float* __restrict__ b, const float* __restrict__ xm,
                            float* __restrict__ out, int total) {
  int i = blockIdx.x * blockDim.x + threadIdx.x;
  if (i >= total) return;
  int f = i % Fn;
  float xd = __expf(-fmaxf(dd[i] * w[f] + b[f], 0.0f));
  float mv = m[i];
  out[i] = mv * x[i] + (1.0f - mv) * xd * x[i] + (1.0f - mv) * (1.0f - xd) * xm[f];
}

// ---------------- GRU-D scan: one block per batch row, 256 threads -------
__global__ void __launch_bounds__(256) grud_scan_kernel(
    const float* __restrict__ xadj, const float* __restrict__ m,
    const float* __restrict__ dd, const float* __restrict__ hdW,
    const float* __restrict__ hdb, const float* __restrict__ Wi,
    const float* __restrict__ bi, const float* __restrict__ Wh,
    const float* __restrict__ bh, const float* __restrict__ h0,
    float* __restrict__ hs) {
  __shared__ float sh_h[256], sh_hh[256], sh_xt[36], sh_dt[18];
  int b = blockIdx.x, t = threadIdx.x;
  sh_h[t] = h0[b * 256 + t];
  __syncthreads();
  for (int step = 0; step < Tn; ++step) {
    long long base = ((long long)b * Tn + step) * Fn;
    if (t < 18)      { sh_dt[t] = dd[base + t]; sh_xt[t] = xadj[base + t]; }
    else if (t < 36) { sh_xt[t] = m[base + t - 18]; }
    __syncthreads();
    float a = hdb[t];
    for (int i = 0; i < 18; ++i) a += sh_dt[i] * hdW[i * 256 + t];
    float hd  = __expf(-fmaxf(a, 0.0f));
    float hhv = hd * sh_h[t];
    sh_hh[t] = hhv;
    __syncthreads();
    float g0 = bi[t], g1 = bi[256 + t], g2 = bi[512 + t];
    for (int i = 0; i < 36; ++i) {
      float xv = sh_xt[i];
      const float* wr = Wi + i * 768;
      g0 += xv * wr[t]; g1 += xv * wr[256 + t]; g2 += xv * wr[512 + t];
    }
    float q0 = bh[t], q1 = bh[256 + t], q2 = bh[512 + t];
    for (int i = 0; i < 256; ++i) {
      float hv = sh_hh[i];
      const float* wr = Wh + i * 768;
      q0 += hv * wr[t]; q1 += hv * wr[256 + t]; q2 += hv * wr[512 + t];
    }
    float r  = 1.0f / (1.0f + __expf(-(g0 + q0)));
    float z  = 1.0f / (1.0f + __expf(-(g1 + q1)));
    float nn = tanhf(g2 + r * q2);
    float hn = (1.0f - z) * nn + z * hhv;
    hs[((long long)b * Tn + step) * 256 + t] = hn;
    __syncthreads();
    sh_h[t] = hn;
    __syncthreads();
  }
}

// ---------------- sinusoidal positional encoding add ---------------------
__global__ void peadd_kernel(float* __restrict__ v, int total) {
  int i = blockIdx.x * blockDim.x + threadIdx.x;
  if (i >= total) return;
  int dpos = i % 256;
  int t = (i / 256) % Tn;
  float div = __expf((float)(dpos & ~1) * (-9.2103403720f / 256.0f));
  float ang = (float)t * div;
  v[i] += (dpos & 1) ? __cosf(ang) : __sinf(ang);
}

// ---------------- patch extraction ----------------
__global__ void __launch_bounds__(256) patch_kernel(const float* __restrict__ img,
                                                    float* __restrict__ patches) {
  int blk = blockIdx.x, j = threadIdx.x;
  int b = blk / SV, p = blk % SV;
  int gy = p / 14, gx = p % 14;
  int py = j / 16, px = j % 16;
  patches[(long long)blk * 256 + j] =
      img[(long long)b * 224 * 224 + (gy * 16 + py) * 224 + gx * 16 + px];
}

// ---------------- broadcast add (vit_pos) ----------------
__global__ void addmod_kernel(float* __restrict__ z, const float* __restrict__ p,
                              int total, int mod) {
  int i = blockIdx.x * blockDim.x + threadIdx.x;
  if (i < total) z[i] += p[i % mod];
}

// ---------------- attention mask from input_lengths ----------------------
__global__ void mask_kernel(const int* __restrict__ len, float* __restrict__ mask) {
  int i = blockIdx.x * blockDim.x + threadIdx.x;
  if (i >= Bn * SF) return;
  int b = i / SF, j = i % SF;
  float v = 0.0f;
  if (j >= 1 && j <= Tn) v = ((j - 1) < len[b]) ? 0.0f : -1e9f;
  mask[i] = v;
}

// ---------------- seq = concat(cls, vslt, img_out) -----------------------
__global__ void concat_kernel(const float* __restrict__ cls, const float* __restrict__ vslt,
                              const float* __restrict__ imgout, float* __restrict__ seq,
                              int total) {
  int i = blockIdx.x * blockDim.x + threadIdx.x;
  if (i >= total) return;
  int d = i % 256;
  int rs = i / 256;
  int b = rs / SF, s = rs % SF;
  float v;
  if (s == 0)        v = cls[d];
  else if (s <= Tn)  v = vslt[((long long)b * Tn + (s - 1)) * 256 + d];
  else               v = imgout[((long long)b * SV + (s - 1 - Tn)) * 256 + d];
  seq[i] = v;
}

// ---------------- per-task heads with batchnorm over batch ---------------
__global__ void __launch_bounds__(64) heads_kernel(
    const float* __restrict__ seq, const float* __restrict__ age,
    const float* __restrict__ gen, const float* __restrict__ W1,
    const float* __restrict__ b1, const float* __restrict__ bng,
    const float* __restrict__ bnb, const float* __restrict__ W2,
    const float* __restrict__ b2, float* __restrict__ out) {
  __shared__ float ci[32][258];
  __shared__ float h1[32][64];
  int k = blockIdx.x, t = threadIdx.x;
  for (int e = t; e < 32 * 258; e += 64) {
    int b = e / 258, dpos = e % 258;
    float v;
    if (dpos < 256)       v = seq[((long long)b * SF) * 256 + dpos];  // seq[b,0,:]
    else if (dpos == 256) v = age[b];
    else                  v = gen[b];
    ci[b][dpos] = v;
  }
  __syncthreads();
  for (int b = 0; b < 32; ++b) {
    float acc = b1[k * 64 + t];
    for (int i = 0; i < 258; ++i)
      acc += ci[b][i] * W1[((long long)k * 258 + i) * 64 + t];
    h1[b][t] = acc;
  }
  float mu = 0.0f;
  for (int b = 0; b < 32; ++b) mu += h1[b][t];
  mu *= (1.0f / 32.0f);
  float var = 0.0f;
  for (int b = 0; b < 32; ++b) { float d = h1[b][t] - mu; var += d * d; }
  var *= (1.0f / 32.0f);
  float is = rsqrtf(var + 1e-5f);
  float gm = bng[k * 64 + t], ba = bnb[k * 64 + t];
  for (int b = 0; b < 32; ++b)
    h1[b][t] = fmaxf((h1[b][t] - mu) * is * gm + ba, 0.0f);
  __syncthreads();
  if (t < 32) {
    float acc = b2[k];
    for (int o = 0; o < 64; ++o) acc += h1[t][o] * W2[k * 64 + o];
    out[k * 32 + t] = 1.0f / (1.0f + __expf(-acc));
  }
}

// ---------------- host orchestration ----------------
struct BlockW {
  const float *W1, *W2, *Wk, *Wo, *Wq, *Wv, *b1, *b2, *bk, *bo, *bq, *bv,
              *ln1b, *ln1g, *ln2b, *ln2g;
};

extern "C" void kernel_launch(void* const* d_in, const int* in_sizes, int n_in,
                              void* d_out, int out_size, void* d_ws, size_t ws_size,
                              hipStream_t stream) {
  (void)in_sizes; (void)n_in; (void)out_size; (void)ws_size;
  auto P = [&](int i) { return (const float*)d_in[i]; };

  // pytree flatten order (dict keys sorted; uppercase sorts before lowercase)
  const float* x   = P(0);  const float* h0  = P(1);  const float* mm = P(2);
  const float* dd  = P(3);  const float* xm  = P(4);  const float* age = P(5);
  const float* gen = P(6);  const int* ilen  = (const int*)d_in[7];
  const float* img = P(8);
  auto getBW = [&](int base) {
    BlockW w;
    w.W1 = P(base + 0);  w.W2 = P(base + 1);  w.Wk = P(base + 2);  w.Wo = P(base + 3);
    w.Wq = P(base + 4);  w.Wv = P(base + 5);  w.b1 = P(base + 6);  w.b2 = P(base + 7);
    w.bk = P(base + 8);  w.bo = P(base + 9);  w.bq = P(base + 10); w.bv = P(base + 11);
    w.ln1b = P(base + 12); w.ln1g = P(base + 13); w.ln2b = P(base + 14); w.ln2g = P(base + 15);
    return w;
  };
  BlockW vitW[6], finW[4];
  for (int i = 0; i < 6; ++i) vitW[i] = getBW(92 + 16 * i);
  for (int i = 0; i < 4; ++i) finW[i] = getBW(16 + 16 * i);

  // ---- workspace layout: bf16 weights first, then f32 scratch ----
  __bf16* wb = (__bf16*)d_ws;
  __bf16* patchWt = wb + 10 * PER_BLK;
  size_t bf_bytes = (size_t)(10 * PER_BLK + 65536) * sizeof(__bf16); // 256B aligned
  float* F0 = (float*)((char*)d_ws + bf_bytes);
  size_t fo = 0;
  auto FA = [&](size_t n) { float* p = F0 + fo; fo += n; return p; };
  size_t padRows = (size_t)ROWPAD * Dn;
  float* xadjb   = FA((size_t)Bn * Tn * Fn);
  float* hsb     = FA((size_t)Bn * Tn * Dn);
  float* vslt    = FA((size_t)Bn * Tn * Dn);
  float* patches = FA((size_t)Bn * SV * Dn);
  float* z       = FA((size_t)Bn * SV * Dn);
  float* seq     = FA((size_t)Bn * SF * Dn);
  float* maskb   = FA((size_t)Bn * SF + 32);
  float* lnbuf   = FA((size_t)Bn * SF * Dn + padRows);
  float* qb      = FA((size_t)Bn * SF * Dn + padRows);
  float* kb      = FA((size_t)Bn * SF * Dn + padRows);
  float* vb      = FA((size_t)Bn * SF * Dn + padRows);
  float* attno   = FA((size_t)Bn * SF * Dn + padRows);
  float* scores  = FA((size_t)Bn * NHh * SP * SP);   // padded, zero-initialized
  float* ffh     = FA((size_t)Bn * SF * FFn);

  auto prepW = [&](const float* W, __bf16* dst, int K, int N) {
    int tot = K * N;
    wprep_kernel<<<(tot + 255) / 256, 256, 0, stream>>>(W, dst, K, N);
  };
  for (int i = 0; i < 10; ++i) {
    const BlockW& w = (i < 6) ? vitW[i] : finW[i - 6];
    __bf16* wbB = wb + (long long)i * PER_BLK;
    prepW(w.Wq, wbB,          256, 256);
    prepW(w.Wk, wbB + 65536,  256, 256);
    prepW(w.Wv, wbB + 131072, 256, 256);
    prepW(w.Wo, wbB + 196608, 256, 256);
    prepW(w.W1, wbB + 262144, 256, 1024);
    prepW(w.W2, wbB + 524288, 1024, 256);
  }
  prepW(P(90), patchWt, 256, 256);

  int scoresN = Bn * NHh * SP * SP;
  zero_kernel<<<(scoresN + 255) / 256, 256, 0, stream>>>(scores, scoresN);

  auto gemmWt = [&](const float* A, const __bf16* Wt, const float* bias,
                    const float* resid, float* C, int M, int N, int K, int act) {
    dim3 g((M + 15) / 16, (N + 127) / 128, 1);
    gemm_wmma_kernel<0><<<g, 128, 0, stream>>>(A, (const void*)Wt, bias, resid, C,
        M, N, K, K, K, N, 1, 0, 0, 0, 0, 0, 0, 1.0f, act);
  };

  auto runBlock = [&](float* xact, int S, const BlockW& w, const __bf16* wbB,
                      int act, const float* maskp) {
    int M = Bn * S;
    const __bf16* WqT = wbB;
    const __bf16* WkT = wbB + 65536;
    const __bf16* WvT = wbB + 131072;
    const __bf16* WoT = wbB + 196608;
    const __bf16* W1T = wbB + 262144;
    const __bf16* W2T = wbB + 524288;
    layernorm256_kernel<<<M, 256, 0, stream>>>(xact, lnbuf, w.ln1g, w.ln1b);
    gemmWt(lnbuf, WqT, w.bq, nullptr, qb, M, 256, 256, 0);
    gemmWt(lnbuf, WkT, w.bk, nullptr, kb, M, 256, 256, 0);
    gemmWt(lnbuf, WvT, w.bv, nullptr, vb, M, 256, 256, 0);
    { // scores[b,h,q,k] = (1/sqrt(dh)) * Q · K^T   (ld = SP, pad stays zero)
      dim3 g((S + 15) / 16, (S + 127) / 128, Bn * NHh);
      gemm_wmma_kernel<1><<<g, 128, 0, stream>>>(qb, (const void*)kb, nullptr, nullptr,
          scores, S, S, DHd, 256, 256, SP, NHh,
          (long long)S * 256, DHd, (long long)S * 256, DHd,
          (long long)NHh * SP * SP, (long long)SP * SP, 0.17677669529f, 0);
    }
    softmax_kernel<<<Bn * NHh * S, 256, 0, stream>>>(scores, maskp, NHh, S);
    { // attno[b,q,h*32+d] = attn · V   (Kpad = SP, zero pad contributes 0)
      dim3 g((S + 15) / 16, 1, Bn * NHh);
      gemm_wmma_kernel<2><<<g, 128, 0, stream>>>(scores, (const void*)vb, nullptr, nullptr,
          attno, S, DHd, SP, SP, 256, 256, NHh,
          (long long)NHh * SP * SP, (long long)SP * SP,
          (long long)S * 256, DHd, (long long)S * 256, DHd, 1.0f, 0);
    }
    gemmWt(attno, WoT, w.bo, xact, xact, M, 256, 256, 0);           // + residual
    layernorm256_kernel<<<M, 256, 0, stream>>>(xact, lnbuf, w.ln2g, w.ln2b);
    gemmWt(lnbuf, W1T, w.b1, nullptr, ffh, M, 1024, 256, act);      // act fused
    gemmWt(ffh, W2T, w.b2, xact, xact, M, 256, 1024, 0);            // + residual
  };

  // ---- pipeline ----
  int totx = Bn * Tn * Fn;
  xadj_kernel<<<(totx + 255) / 256, 256, 0, stream>>>(x, mm, dd, P(87), P(86), xm,
                                                      xadjb, totx);
  grud_scan_kernel<<<Bn, 256, 0, stream>>>(xadjb, mm, dd, P(84), P(85), P(81), P(83),
                                           P(80), P(82), h0, hsb);
  layernorm256_kernel<<<Bn * Tn, 256, 0, stream>>>(hsb, vslt, P(89), P(88));
  peadd_kernel<<<(Bn * Tn * Dn + 255) / 256, 256, 0, stream>>>(vslt, Bn * Tn * Dn);

  patch_kernel<<<Bn * SV, 256, 0, stream>>>(img, patches);
  gemmWt(patches, patchWt, P(91), nullptr, z, Bn * SV, 256, 256, 0);
  addmod_kernel<<<(Bn * SV * Dn + 255) / 256, 256, 0, stream>>>(z, P(190),
                                                               Bn * SV * Dn, SV * Dn);
  for (int i = 0; i < 6; ++i)
    runBlock(z, SV, vitW[i], wb + (long long)i * PER_BLK, /*gelu*/2, nullptr);
  layernorm256_kernel<<<Bn * SV, 256, 0, stream>>>(z, lnbuf, P(189), P(188));

  mask_kernel<<<(Bn * SF + 255) / 256, 256, 0, stream>>>(ilen, maskb);
  concat_kernel<<<(Bn * SF * Dn + 255) / 256, 256, 0, stream>>>(P(11), vslt, lnbuf,
                                                                seq, Bn * SF * Dn);
  for (int i = 0; i < 4; ++i)
    runBlock(seq, SF, finW[i], wb + (long long)(6 + i) * PER_BLK, /*relu*/1, maskb);

  heads_kernel<<<NTk, 64, 0, stream>>>(seq, age, gen, P(12), P(14), P(10), P(9),
                                       P(13), P(15), (float*)d_out);
}
